// SparseAttention_65292092834435
// MI455X (gfx1250) — matrix operand order repaired and checked by reading, not compiled, not run on hardware
//
#include <hip/hip_runtime.h>
#include <hip/hip_bf16.h>

// ---------------- problem constants ----------------
constexpr int Sq = 4096;   // sequence length
constexpr int Dm = 1024;   // model dim
constexpr int NH = 16;     // heads
constexpr int HD = 64;     // head dim
constexpr int W2 = 64;     // half window (|i-j| <= 64)

typedef __attribute__((ext_vector_type(16))) __bf16 v16bf;
typedef __attribute__((ext_vector_type(8)))  float  v8f;

union Frag16 {            // 16 bf16 = 32 bytes = two uint4
    uint4 u[2];
    v16bf v;
};

__device__ __forceinline__ __hip_bfloat16 f2bf(float f) { return __float2bfloat16(f); }

// ---- CDNA5 async global->LDS copy (ASYNCcnt path), guarded for portability ----
#if defined(__HIP_DEVICE_COMPILE__) && defined(__gfx1250__) && \
    __has_builtin(__builtin_amdgcn_global_load_async_to_lds_b128) && \
    __has_builtin(__builtin_amdgcn_s_wait_asynccnt)
#define HAVE_ASYNC_LDS 1
#else
#define HAVE_ASYNC_LDS 0
#endif

#if HAVE_ASYNC_LDS
typedef int vi4 __attribute__((vector_size(16)));              // int4 vector
typedef __attribute__((address_space(1))) vi4 glb_vi4;         // global int4
typedef __attribute__((address_space(3))) vi4 lds_vi4;         // LDS int4
#endif

// copy 16 bytes global -> LDS (async on gfx1250, sync fallback elsewhere)
__device__ __forceinline__ void cp16(void* lds_dst, const void* glb_src) {
#if HAVE_ASYNC_LDS
    __builtin_amdgcn_global_load_async_to_lds_b128(
        (glb_vi4*)glb_src, (lds_vi4*)lds_dst, 0, 0);
#else
    *(uint4*)lds_dst = *(const uint4*)glb_src;
#endif
}
template <int N>
__device__ __forceinline__ void cp_wait_n() {
#if HAVE_ASYNC_LDS
    __builtin_amdgcn_s_wait_asynccnt(N);   // s_wait_asynccnt N
#endif
}

// ---------------- pack kernels ----------------
__global__ void cvt_bf16_kernel(const float* __restrict__ src,
                                __hip_bfloat16* __restrict__ dst, int n) {
    int i = blockIdx.x * 256 + threadIdx.x;
    if (i < n) dst[i] = f2bf(src[i]);
}

// Pack torch Linear weight W[N_out, K_in] into per-(BKxBN)-tile B layout:
// tile (k0/32, n0/128) is a contiguous 128x32 block stored [n][k], so GEMM
// B staging is a flat contiguous copy (no LDS-side transpose).
__global__ void pack_w_kernel(const float* __restrict__ W,
                              __hip_bfloat16* __restrict__ Wp,
                              int N, int K) {
    int i = blockIdx.x * 256 + threadIdx.x;
    if (i < N * K) {
        int n = i / K, k = i % K;   // coalesced read over k
        size_t o = ((size_t)(k >> 5) * (size_t)(N >> 7) + (size_t)(n >> 7)) * 4096
                 + (size_t)(n & 127) * 32 + (size_t)(k & 31);
        Wp[o] = f2bf(W[i]);
    }
}

// ---------------- WMMA GEMM: C = (A @ W^T + bias) * scale ----------------
// A     : [M, Kd] bf16 row-major
// Bpack : pre-packed tiles (see pack_w_kernel)
// cmode : 0 -> Cf f32 [M,N]; 1 -> Ch bf16 [M,N]; 2 -> Ch bf16 transposed [N,M]
// Double-buffered LDS: async copies for tile i+1 are in flight while tile i
// feeds the WMMAs; s_wait_asynccnt 4 fences only the older buffer's 4 copies.
constexpr int BM = 128, BN = 128, BK = 32;

__global__ __launch_bounds__(256)
void gemm_bf16_kernel(const __hip_bfloat16* __restrict__ A,
                      const __hip_bfloat16* __restrict__ Bpack,
                      const float* __restrict__ bias,
                      float* __restrict__ Cf,
                      __hip_bfloat16* __restrict__ Ch,
                      int M, int N, int Kd, float scale, int cmode) {
    __shared__ alignas(16) __hip_bfloat16 As[2][BM][BK];    // 16 KB ping-pong
    __shared__ alignas(16) __hip_bfloat16 Bst[2][BN][BK];   // 16 KB ping-pong

    const int tid  = threadIdx.x;
    const int lane = tid & 31;
    const int wave = tid >> 5;                 // 8 waves
    const int wm   = (wave >> 1) * 32;         // wave row block
    const int wn   = (wave & 1) * 64;          // wave col block
    const int m0   = blockIdx.y * BM;
    const int n0   = blockIdx.x * BN;
    const int half = lane >> 4;
    const int l15  = lane & 15;

    v8f acc[2][4];
    #pragma unroll
    for (int i = 0; i < 2; ++i)
        #pragma unroll
        for (int j = 0; j < 4; ++j) {
            v8f z = {0.f, 0.f, 0.f, 0.f, 0.f, 0.f, 0.f, 0.f};
            acc[i][j] = z;
        }

    const int ar   = tid >> 1;                  // A staging row
    const int aseg = (tid & 1) * 16;            // A staging col segment

    auto stage = [&](int k0, int buf) {         // 4 async b128 per thread
        cp16(&As[buf][ar][aseg],
             &A[(size_t)(m0 + ar) * Kd + k0 + aseg]);
        cp16(&As[buf][ar][aseg + 8],
             &A[(size_t)(m0 + ar) * Kd + k0 + aseg + 8]);
        const __hip_bfloat16* bsrc =
            Bpack + ((size_t)(k0 >> 5) * (size_t)(N >> 7) + (size_t)(n0 >> 7)) * 4096;
        cp16((char*)&Bst[buf][0][0] + tid * 32, (const char*)bsrc + tid * 32);
        cp16((char*)&Bst[buf][0][0] + tid * 32 + 16, (const char*)bsrc + tid * 32 + 16);
    };

    const int nk = Kd / BK;
    stage(0, 0);                                // prologue: fill buffer 0

    for (int ki = 0; ki < nk; ++ki) {
        const int cur = ki & 1;
        if (ki + 1 < nk) {
            stage((ki + 1) * BK, cur ^ 1);      // overlap next tile's copies
            cp_wait_n<4>();                     // fence only the older 4 copies
        } else {
            cp_wait_n<0>();
        }
        __syncthreads();

        // ---- A fragments (16x32 bf16; lane holds row l15, K = {b..b+7, 16+b..+7})
        Frag16 af[2];
        const int abase = half * 8;
        #pragma unroll
        for (int mt = 0; mt < 2; ++mt) {
            int r = wm + mt * 16 + l15;
            af[mt].u[0] = *reinterpret_cast<const uint4*>(&As[cur][r][abase]);
            af[mt].u[1] = *reinterpret_cast<const uint4*>(&As[cur][r][16 + abase]);
        }
        // ---- B fragments (32x16 bf16; lane holds col l15, K = {b..b+7, 16+b..+7})
        #pragma unroll
        for (int nt = 0; nt < 4; ++nt) {
            Frag16 bfg;
            int c = wn + nt * 16 + l15;
            int koff = half * 8;
            bfg.u[0] = *reinterpret_cast<const uint4*>(&Bst[cur][c][koff]);
            bfg.u[1] = *reinterpret_cast<const uint4*>(&Bst[cur][c][koff + 16]);
            #pragma unroll
            for (int mt = 0; mt < 2; ++mt) {
                acc[mt][nt] = __builtin_amdgcn_wmma_f32_16x16x32_bf16(
                    false, af[mt].v, false, bfg.v, (short)0, acc[mt][nt],
                    false, false);
            }
        }
        __syncthreads();   // all reads done before next iter overwrites buffer
    }

    // ---- epilogue: C-frag row = half*8+v, col = l15
    #pragma unroll
    for (int mt = 0; mt < 2; ++mt)
        #pragma unroll
        for (int nt = 0; nt < 4; ++nt)
            #pragma unroll
            for (int v = 0; v < 8; ++v) {
                int gr = m0 + wm + mt * 16 + half * 8 + v;
                int gc = n0 + wn + nt * 16 + l15;
                float val = (acc[mt][nt][v] + bias[gc]) * scale;
                if (cmode == 0)      Cf[(size_t)gr * N + gc] = val;
                else if (cmode == 1) Ch[(size_t)gr * N + gc] = f2bf(val);
                else                 Ch[(size_t)gc * M + gr] = f2bf(val);  // [N,M]
            }
}

// ---------------- windowed attention ----------------
// grid: (NH, Sq/64), block: 128 threads (4 waves, 16 query rows per wave)
// Qb pre-scaled by 1/sqrt(HD). VtG is the V projection stored transposed [D, S].
__global__ __launch_bounds__(128)
void attn_kernel(const __hip_bfloat16* __restrict__ Qb,
                 const __hip_bfloat16* __restrict__ Kb,
                 const __hip_bfloat16* __restrict__ VtG,
                 __hip_bfloat16* __restrict__ Ctx) {
    __shared__ alignas(16) __hip_bfloat16 Kl[192][64];    // 24 KB  key stripe
    __shared__ alignas(16) __hip_bfloat16 Vt[64][192];    // 24 KB  V transposed
    __shared__ alignas(16) __hip_bfloat16 Pl[4][16][32];  //  4 KB  per-wave P tile

    const int h   = blockIdx.x;
    const int q0  = blockIdx.y * 64;
    const int ks0 = q0 - 64;                 // first staged key (may be <0)
    const int tid = threadIdx.x;
    const int lane = tid & 31, wave = tid >> 5;
    const int half = lane >> 4, l15 = lane & 15;

    // stage K stripe row-major: 192 rows x 64 cols, 8-bf16 chunks
    for (int vi = tid; vi < 192 * 64 / 8; vi += 128) {
        int j = vi >> 3, seg = (vi & 7) * 8;
        int key = ks0 + j;
        if (key >= 0 && key < Sq)
            cp16(&Kl[j][seg], &Kb[(size_t)key * Dm + h * HD + seg]);
        else
            *reinterpret_cast<uint4*>(&Kl[j][seg]) = make_uint4(0u, 0u, 0u, 0u);
    }
    // stage V transposed from VtG: row c contiguous over keys, 8-bf16 chunks
    for (int ci = tid; ci < 64 * 24; ci += 128) {
        int c = ci / 24, jc = (ci % 24) * 8;
        int k0 = ks0 + jc;
        if (k0 >= 0 && k0 + 8 <= Sq) {
            cp16(&Vt[c][jc], &VtG[(size_t)(h * HD + c) * Sq + k0]);
        } else {
            for (int e = 0; e < 8; ++e) {
                int key = k0 + e;
                Vt[c][jc + e] = (key >= 0 && key < Sq)
                                    ? VtG[(size_t)(h * HD + c) * Sq + key]
                                    : f2bf(0.0f);
            }
        }
    }
    cp_wait_n<0>();
    __syncthreads();

    const int qw = q0 + wave * 16;
    const int abase = half * 8;

    // Q A-fragments for hd chunks 0..31 and 32..63 (straight from L2)
    Frag16 aQ[2];
    {
        const __hip_bfloat16* qp = &Qb[(size_t)(qw + l15) * Dm + h * HD];
        #pragma unroll
        for (int ch = 0; ch < 2; ++ch) {
            aQ[ch].u[0] = *reinterpret_cast<const uint4*>(qp + ch * 32 + abase);
            aQ[ch].u[1] = *reinterpret_cast<const uint4*>(qp + ch * 32 + 16 + abase);
        }
    }

    auto score_tile = [&](int t) -> v8f {
        v8f c = {0.f, 0.f, 0.f, 0.f, 0.f, 0.f, 0.f, 0.f};
        int col  = t * 16 + l15;          // key column within stripe
        int koff = half * 8;              // hd slice within chunk
        Frag16 bK0, bK1;
        bK0.u[0] = *reinterpret_cast<const uint4*>(&Kl[col][koff]);
        bK0.u[1] = *reinterpret_cast<const uint4*>(&Kl[col][koff + 16]);
        bK1.u[0] = *reinterpret_cast<const uint4*>(&Kl[col][32 + koff]);
        bK1.u[1] = *reinterpret_cast<const uint4*>(&Kl[col][32 + koff + 16]);
        c = __builtin_amdgcn_wmma_f32_16x16x32_bf16(false, aQ[0].v, false, bK0.v,
                                                    (short)0, c, false, false);
        c = __builtin_amdgcn_wmma_f32_16x16x32_bf16(false, aQ[1].v, false, bK1.v,
                                                    (short)0, c, false, false);
        return c;
    };

    // ---- pass 1: row max over masked scores
    float rmax[8];
    #pragma unroll
    for (int v = 0; v < 8; ++v) rmax[v] = -3.0e38f;
    for (int t = 0; t < 12; ++t) {
        v8f c = score_tile(t);
        int kg = ks0 + t * 16 + l15;
        #pragma unroll
        for (int v = 0; v < 8; ++v) {
            int qg = qw + half * 8 + v;
            bool ok = (kg >= 0) && (kg < Sq) && (qg - kg <= W2) && (kg - qg <= W2);
            rmax[v] = fmaxf(rmax[v], ok ? c[v] : -3.0e38f);
        }
    }
    #pragma unroll
    for (int m = 1; m < 16; m <<= 1)
        #pragma unroll
        for (int v = 0; v < 8; ++v)
            rmax[v] = fmaxf(rmax[v], __shfl_xor(rmax[v], m, 32));

    // ---- pass 2: exp, row sums, P @ V in 32-key chunks
    float rsum[8];
    #pragma unroll
    for (int v = 0; v < 8; ++v) rsum[v] = 0.f;
    v8f acc[4];
    #pragma unroll
    for (int nt = 0; nt < 4; ++nt) {
        v8f z = {0.f, 0.f, 0.f, 0.f, 0.f, 0.f, 0.f, 0.f};
        acc[nt] = z;
    }

    for (int cc = 0; cc < 6; ++cc) {
        #pragma unroll
        for (int s2 = 0; s2 < 2; ++s2) {
            int t = cc * 2 + s2;
            v8f c = score_tile(t);
            int kg = ks0 + t * 16 + l15;
            #pragma unroll
            for (int v = 0; v < 8; ++v) {
                int qg = qw + half * 8 + v;
                bool ok = (kg >= 0) && (kg < Sq) && (qg - kg <= W2) && (kg - qg <= W2);
                float p = ok ? __expf(c[v] - rmax[v]) : 0.0f;
                rsum[v] += p;
                Pl[wave][half * 8 + v][s2 * 16 + l15] = f2bf(p);
            }
        }
        // P A-fragment (same-wave LDS RAW; DS ops are in-order per wave)
        Frag16 aP;
        aP.u[0] = *reinterpret_cast<const uint4*>(&Pl[wave][l15][abase]);
        aP.u[1] = *reinterpret_cast<const uint4*>(&Pl[wave][l15][16 + abase]);

        int koff = cc * 32 + half * 8;
        #pragma unroll
        for (int nt = 0; nt < 4; ++nt) {
            Frag16 bV;
            int d = nt * 16 + l15;
            bV.u[0] = *reinterpret_cast<const uint4*>(&Vt[d][koff]);
            bV.u[1] = *reinterpret_cast<const uint4*>(&Vt[d][koff + 16]);
            acc[nt] = __builtin_amdgcn_wmma_f32_16x16x32_bf16(
                false, aP.v, false, bV.v, (short)0, acc[nt], false, false);
        }
    }

    #pragma unroll
    for (int m = 1; m < 16; m <<= 1)
        #pragma unroll
        for (int v = 0; v < 8; ++v) rsum[v] += __shfl_xor(rsum[v], m, 32);

    #pragma unroll
    for (int nt = 0; nt < 4; ++nt)
        #pragma unroll
        for (int v = 0; v < 8; ++v) {
            int qg = qw + half * 8 + v;
            int gc = h * HD + nt * 16 + l15;
            Ctx[(size_t)qg * Dm + gc] = f2bf(acc[nt][v] / rsum[v]);
        }
}

// ---------------- launcher ----------------
extern "C" void kernel_launch(void* const* d_in, const int* in_sizes, int n_in,
                              void* d_out, int out_size, void* d_ws, size_t ws_size,
                              hipStream_t stream) {
    const float* x  = (const float*)d_in[0];
    const float* Wq = (const float*)d_in[1];
    const float* bq = (const float*)d_in[2];
    const float* Wk = (const float*)d_in[3];
    const float* bk = (const float*)d_in[4];
    const float* Wv = (const float*)d_in[5];
    const float* bv = (const float*)d_in[6];
    const float* Wo = (const float*)d_in[7];
    const float* bo = (const float*)d_in[8];

    char* ws = (char*)d_ws;
    size_t off = 0;
    auto alloc_bf = [&](size_t elems) {
        __hip_bfloat16* p = (__hip_bfloat16*)(ws + off);
        off += elems * sizeof(__hip_bfloat16);
        return p;
    };
    __hip_bfloat16* xb   = alloc_bf((size_t)Sq * Dm);
    __hip_bfloat16* WqP  = alloc_bf((size_t)Dm * Dm);
    __hip_bfloat16* WkP  = alloc_bf((size_t)Dm * Dm);
    __hip_bfloat16* WvP  = alloc_bf((size_t)Dm * Dm);
    __hip_bfloat16* WoP  = alloc_bf((size_t)Dm * Dm);
    __hip_bfloat16* Qb   = alloc_bf((size_t)Sq * Dm);
    __hip_bfloat16* Kb   = alloc_bf((size_t)Sq * Dm);
    __hip_bfloat16* VtG  = alloc_bf((size_t)Sq * Dm);   // transposed [D, S]
    __hip_bfloat16* Ctxb = alloc_bf((size_t)Sq * Dm);

    // pack inputs to bf16 / tiled layouts
    {
        int n = Sq * Dm;
        cvt_bf16_kernel<<<(n + 255) / 256, 256, 0, stream>>>(x, xb, n);
        int nw = Dm * Dm;
        pack_w_kernel<<<(nw + 255) / 256, 256, 0, stream>>>(Wq, WqP, Dm, Dm);
        pack_w_kernel<<<(nw + 255) / 256, 256, 0, stream>>>(Wk, WkP, Dm, Dm);
        pack_w_kernel<<<(nw + 255) / 256, 256, 0, stream>>>(Wv, WvP, Dm, Dm);
        pack_w_kernel<<<(nw + 255) / 256, 256, 0, stream>>>(Wo, WoP, Dm, Dm);
    }

    dim3 ggrid(Dm / BN, Sq / BM);   // (8, 32)
    const float qscale = 0.125f;    // 1/sqrt(HD), folded into Q
    gemm_bf16_kernel<<<ggrid, 256, 0, stream>>>(xb, WqP, bq, nullptr, Qb,
                                                Sq, Dm, Dm, qscale, 1);
    gemm_bf16_kernel<<<ggrid, 256, 0, stream>>>(xb, WkP, bk, nullptr, Kb,
                                                Sq, Dm, Dm, 1.0f, 1);
    gemm_bf16_kernel<<<ggrid, 256, 0, stream>>>(xb, WvP, bv, nullptr, VtG,
                                                Sq, Dm, Dm, 1.0f, 2);

    dim3 agrid(NH, Sq / 64);        // (16, 64)
    attn_kernel<<<agrid, 128, 0, stream>>>(Qb, Kb, VtG, Ctxb);

    gemm_bf16_kernel<<<ggrid, 256, 0, stream>>>(Ctxb, WoP, bo, (float*)d_out,
                                                nullptr, Sq, Dm, Dm, 1.0f, 0);
}
